// NemotronHMamba2Mixer_2216203125386
// MI455X (gfx1250) — matrix operand order repaired and checked by reading, not compiled, not run on hardware
//
#include <hip/hip_runtime.h>
#include <hip/hip_bf16.h>

// ---------------------------------------------------------------------------
// NemotronH Mamba2 mixer for gfx1250 (MI455X), bf16 WMMA (f32 accumulate).
// GEMMs: L2-friendly stripe ordering + async-to-LDS double-buffered weights.
// ---------------------------------------------------------------------------

#define H_SIZE   4096
#define NH       128
#define HD       64
#define SS       128
#define CHUNK    128
#define NC       16          // 2048 / 128
#define SEQ      2048
#define NG       8
#define INTER    8192
#define CONV_DIM 10240
#define PROJ     18560

typedef __attribute__((ext_vector_type(16))) __bf16 v16bf;
typedef __attribute__((ext_vector_type(8)))  float  v8f;

static __device__ inline v8f wmma_bf16(v16bf a, v16bf b, v8f c) {
    // D = A(16x32 bf16) * B(32x16 bf16) + C(16x16 f32)
    return __builtin_amdgcn_wmma_f32_16x16x32_bf16(
        false, a, false, b, (short)0, c, false, false);
}

// ---- LDS fragment loaders (layouts per CDNA5 ISA 7.12.2, wave32) ----------

// A operand, A[m][k] = s[m*stride + k], K-window k0..k0+31.
static __device__ inline v16bf ldsA(const __bf16* s, int stride, int m0, int k0, int lane) {
    const __bf16* r = s + (m0 + (lane & 15)) * stride + k0;
    int g = (lane >> 4) * 8;
    v16bf a;
#pragma unroll
    for (int h = 0; h < 8; ++h) a[h] = r[g + h];
#pragma unroll
    for (int h = 0; h < 8; ++h) a[8 + h] = r[16 + g + h];
    return a;
}

// A operand where A[m][k] = s[k*stride + m] * scale[k]  (transposed + scaled)
static __device__ inline v16bf ldsA_t_scaled(const __bf16* s, int stride, int m0, int k0,
                                             int lane, const float* scale) {
    int m = m0 + (lane & 15);
    int g = (lane >> 4) * 8;
    v16bf a;
#pragma unroll
    for (int h = 0; h < 8; ++h) {
        int k = k0 + g + h;
        a[h] = (__bf16)((float)s[k * stride + m] * scale[k]);
    }
#pragma unroll
    for (int h = 0; h < 8; ++h) {
        int k = k0 + 16 + g + h;
        a[8 + h] = (__bf16)((float)s[k * stride + m] * scale[k]);
    }
    return a;
}

// B operand, B'[k][n] = s[n*stride + k]   (lane reads two 8-elem runs, table order)
static __device__ inline v16bf ldsB_rows(const __bf16* s, int stride, int n0, int k0, int lane) {
    const __bf16* r = s + (n0 + (lane & 15)) * stride + k0 + (lane >> 4) * 8;
    v16bf b;
#pragma unroll
    for (int h = 0; h < 16; ++h) b[h] = r[h < 8 ? h : h + 8];
    return b;
}

// B operand, B'[k][n] = s[k*stride + n]   (strided LDS reads)
static __device__ inline v16bf ldsB_cols(const __bf16* s, int stride, int n0, int k0, int lane) {
    int n = n0 + (lane & 15);
    int kk = k0 + (lane >> 4) * 8;
    v16bf b;
#pragma unroll
    for (int h = 0; h < 16; ++h) b[h] = s[(kk + (h < 8 ? h : h + 8)) * stride + n];
    return b;
}

static __device__ inline void pk8(v16bf& v, int o, float4 a, float4 b) {
    v[o + 0] = (__bf16)a.x; v[o + 1] = (__bf16)a.y;
    v[o + 2] = (__bf16)a.z; v[o + 3] = (__bf16)a.w;
    v[o + 4] = (__bf16)b.x; v[o + 5] = (__bf16)b.y;
    v[o + 6] = (__bf16)b.z; v[o + 7] = (__bf16)b.w;
}

// ---------------------------------------------------------------------------
// GEMM: C[M,N] = A[M,K] @ W[N,K]^T  (f32 in/out, bf16 WMMA inside)
// Block = 8 waves sharing one 64-wide N-stripe of W; each wave owns a 64x64
// output tile. W stripe (64 x 32 f32 per K-step) is DMA'd once per block into
// LDS with GLOBAL_LOAD_ASYNC_TO_LDS_B128 (double-buffered, ASYNCcnt), A is
// streamed per-wave from global (L2-resident across stripes).
// Grid: x = (N/64)*4 ; nb = blockIdx.x>>2 (stripe), mb = blockIdx.x&3.
// Requires M==2048, N%64==0, K%32==0.
// ---------------------------------------------------------------------------
__global__ __launch_bounds__(256)
void gemm_bf16w(const float* __restrict__ A, const float* __restrict__ W,
                float* __restrict__ C, int M, int N, int K) {
    __shared__ float sW[2][64 * 32];
    const int tid = threadIdx.x;
    const int lane = tid & 31;
    const int wave = tid >> 5;
    const int nb = blockIdx.x >> 2;
    const int mb = blockIdx.x & 3;
    const int n0 = nb * 64;
    const int m0 = (mb * 8 + wave) * 64;
    const int nsteps = K >> 5;

    // async-copy one 64x32 f32 W stripe into sW[buf]; 512 16B chunks / block
    auto issue = [&](int s) {
        const int k0 = s << 5;
        const int buf = s & 1;
#pragma unroll
        for (int u = 0; u < 2; ++u) {
            int i = tid + u * 256;            // chunk id 0..511
            int row = i >> 3;                 // n-row within stripe
            int c4 = (i & 7) * 4;             // f32 column within stripe
            unsigned ldsoff = (unsigned)(size_t)(&sW[buf][i * 4]);
            unsigned long long ga =
                (unsigned long long)(size_t)(W + (size_t)(n0 + row) * K + k0 + c4);
            asm volatile("global_load_async_to_lds_b128 %0, %1, off"
                         :: "v"(ldsoff), "v"(ga) : "memory");
        }
    };

    issue(0);

    v8f acc[4][4];
#pragma unroll
    for (int i = 0; i < 4; ++i)
#pragma unroll
        for (int j = 0; j < 4; ++j) acc[i][j] = (v8f){0.f,0.f,0.f,0.f,0.f,0.f,0.f,0.f};

    const int mA = lane & 15;
    const int g8 = (lane >> 4) * 8;
    const int g16 = (lane >> 4) * 16;

    for (int s = 0; s < nsteps; ++s) {
        const int k0 = s << 5;
        const int buf = s & 1;
        asm volatile("s_wait_asynccnt 0" ::: "memory");
        __syncthreads();
        if (s + 1 < nsteps) issue(s + 1);

        v16bf b[4];
#pragma unroll
        for (int j = 0; j < 4; ++j) {
            const float* r = &sW[buf][(j * 16 + mA) * 32 + g8];
#pragma unroll
            for (int h = 0; h < 16; ++h) b[j][h] = (__bf16)r[h < 8 ? h : h + 8];
        }
#pragma unroll
        for (int i = 0; i < 4; ++i) {
            v16bf a;
            const float4* r = reinterpret_cast<const float4*>(
                A + (size_t)(m0 + i * 16 + mA) * K + k0);
            pk8(a, 0, r[g8 >> 2], r[(g8 >> 2) + 1]);
            pk8(a, 8, r[(16 + g8) >> 2], r[((16 + g8) >> 2) + 1]);
#pragma unroll
            for (int j = 0; j < 4; ++j) acc[i][j] = wmma_bf16(a, b[j], acc[i][j]);
        }
        __syncthreads();   // everyone done reading buf before it is re-filled
    }

#pragma unroll
    for (int i = 0; i < 4; ++i)
#pragma unroll
        for (int j = 0; j < 4; ++j) {
            int n = n0 + j * 16 + (lane & 15);
#pragma unroll
            for (int r = 0; r < 8; ++r) {
                int m = m0 + i * 16 + r + 8 * (lane >> 4);
                C[(size_t)m * N + n] = acc[i][j][r];
            }
        }
}

// ---------------------------------------------------------------------------
// Causal depthwise conv1d (K=4) + bias + SiLU on xBC columns of proj.
// ---------------------------------------------------------------------------
__global__ __launch_bounds__(256)
void conv_silu_kernel(const float* __restrict__ proj, const float* __restrict__ cw,
                      const float* __restrict__ cb, float* __restrict__ xBCs) {
    int idx = blockIdx.x * 256 + threadIdx.x;
    if (idx >= SEQ * CONV_DIM) return;
    int t = idx / CONV_DIM, c = idx % CONV_DIM;
    float acc = cb[c];
#pragma unroll
    for (int j = 0; j < 4; ++j) {
        int ts = t - 3 + j;
        float v = (ts >= 0) ? proj[(size_t)ts * PROJ + INTER + c] : 0.f;
        acc = fmaf(v, cw[c * 4 + j], acc);
    }
    xBCs[idx] = acc / (1.f + __expf(-acc));
}

// ---------------------------------------------------------------------------
// SSD stage 1: per (chunk, head) -> dt/cumsum, chunk end-states, masked
// attention scores, intra-chunk output (Ydiag + D skip). 256 threads = 8 waves.
// LDS ~57KB: sB 32K + sX 16K + sCt 8K + scalars.
// ---------------------------------------------------------------------------
__global__ __launch_bounds__(256)
void ssd1_kernel(const float* __restrict__ proj, const float* __restrict__ xBCs,
                 const float* __restrict__ dt_bias, const float* __restrict__ Dp,
                 float* __restrict__ Ybuf, float* __restrict__ states,
                 float* __restrict__ csbuf, float* __restrict__ cdbuf) {
    __shared__ __bf16 sB[128 * 128];   // Bm chunk [l][n]; reused for masked scores [l][s]
    __shared__ __bf16 sX[128 * 64];    // xdt [l][p]
    __shared__ __bf16 sCt[128 * 32];   // staged C K-slice
    __shared__ float s_dtc[128], s_cs[128], s_decay[128];

    const int c = blockIdx.x >> 7;
    const int h = blockIdx.x & 127;
    const int tid = threadIdx.x, lane = tid & 31, wave = tid >> 5;
    const int grp = h >> 4;
    const float Ah = -(float)(h + 1);

    if (tid < 128) {
        float raw = proj[(size_t)(c * 128 + tid) * PROJ + (INTER + CONV_DIM) + h] + dt_bias[h];
        float sp = raw > 20.f ? raw : log1pf(__expf(raw));
        s_dtc[tid] = fminf(fmaxf(sp, 0.001f), 100.f);
    }
    __syncthreads();
    if (tid == 0) {
        float a = 0.f;
        for (int l = 0; l < 128; ++l) { a += s_dtc[l] * Ah; s_cs[l] = a; }
    }
    __syncthreads();
    if (tid < 128) s_decay[tid] = __expf(s_cs[127] - s_cs[tid]);

    for (int i = tid; i < 128 * 128; i += 256) {
        int l = i >> 7, n = i & 127;
        sB[i] = (__bf16)xBCs[(size_t)(c * 128 + l) * CONV_DIM + INTER + grp * 128 + n];
    }
    for (int i = tid; i < 128 * 64; i += 256) {
        int l = i >> 6, p = i & 63;
        sX[i] = (__bf16)(xBCs[(size_t)(c * 128 + l) * CONV_DIM + h * 64 + p] * s_dtc[l]);
    }
    __syncthreads();

    // ---- chunk end-states: st[p][n] = sum_l xdt[l][p] * decay[l] * B[l][n] ----
    {
        int mt = wave >> 1;
        int nbase = (wave & 1) * 4;
        v8f acc[4];
#pragma unroll
        for (int j = 0; j < 4; ++j) acc[j] = (v8f){0.f,0.f,0.f,0.f,0.f,0.f,0.f,0.f};
        for (int k0 = 0; k0 < 128; k0 += 32) {
            v16bf a = ldsA_t_scaled(sX, 64, mt * 16, k0, lane, s_decay);
#pragma unroll
            for (int j = 0; j < 4; ++j) {
                v16bf b = ldsB_cols(sB, 128, (nbase + j) * 16, k0, lane);
                acc[j] = wmma_bf16(a, b, acc[j]);
            }
        }
        size_t base = (size_t)(c * 128 + h) * 64;
#pragma unroll
        for (int j = 0; j < 4; ++j) {
            int n = (nbase + j) * 16 + (lane & 15);
#pragma unroll
            for (int r = 0; r < 8; ++r) {
                int p = mt * 16 + r + 8 * (lane >> 4);
                states[(base + p) * 128 + n] = acc[j][r];
            }
        }
    }

    // ---- scores[l][s] = sum_n C[l][n] * B[s][n]  (C staged 128x32 slices) ----
    v8f sc[8];
#pragma unroll
    for (int j = 0; j < 8; ++j) sc[j] = (v8f){0.f,0.f,0.f,0.f,0.f,0.f,0.f,0.f};
    for (int k0 = 0; k0 < 128; k0 += 32) {
        __syncthreads();
        for (int i = tid; i < 128 * 32; i += 256) {
            int l = i >> 5, n = i & 31;
            sCt[i] = (__bf16)xBCs[(size_t)(c * 128 + l) * CONV_DIM + INTER + NG * SS
                                  + grp * 128 + k0 + n];
        }
        __syncthreads();
        v16bf a = ldsA(sCt, 32, wave * 16, 0, lane);
#pragma unroll
        for (int j = 0; j < 8; ++j) {
            v16bf b = ldsB_rows(sB, 128, j * 16, k0, lane);
            sc[j] = wmma_bf16(a, b, sc[j]);
        }
    }
    __syncthreads();
#pragma unroll
    for (int j = 0; j < 8; ++j) {
        int s = j * 16 + (lane & 15);
#pragma unroll
        for (int r = 0; r < 8; ++r) {
            int l = wave * 16 + r + 8 * (lane >> 4);
            float v = (l >= s) ? sc[j][r] * __expf(s_cs[l] - s_cs[s]) : 0.f;
            sB[l * 128 + s] = (__bf16)v;
        }
    }
    __syncthreads();

    // ---- Ydiag[l][p] = scores @ xdt  (+ D*x skip) ----
    {
        v8f acc[4];
#pragma unroll
        for (int j = 0; j < 4; ++j) acc[j] = (v8f){0.f,0.f,0.f,0.f,0.f,0.f,0.f,0.f};
        for (int k0 = 0; k0 < 128; k0 += 32) {
            v16bf a = ldsA(sB, 128, wave * 16, k0, lane);
#pragma unroll
            for (int j = 0; j < 4; ++j) {
                v16bf b = ldsB_cols(sX, 64, j * 16, k0, lane);
                acc[j] = wmma_bf16(a, b, acc[j]);
            }
        }
        float Dh = Dp[h];
#pragma unroll
        for (int j = 0; j < 4; ++j) {
            int p = j * 16 + (lane & 15);
#pragma unroll
            for (int r = 0; r < 8; ++r) {
                int l = wave * 16 + r + 8 * (lane >> 4);
                size_t gi = (size_t)(c * 128 + l) * INTER + h * 64 + p;
                float xv = xBCs[(size_t)(c * 128 + l) * CONV_DIM + h * 64 + p];
                Ybuf[gi] = acc[j][r] + Dh * xv;
            }
        }
    }
    if (tid < 128) csbuf[(size_t)(c * 128 + h) * 128 + tid] = s_cs[tid];
    if (tid == 0)  cdbuf[c * 128 + h] = __expf(s_cs[127]);
}

// ---------------------------------------------------------------------------
// SSD stage 2: inter-chunk recurrence (16 steps), in-place states -> prev.
// ---------------------------------------------------------------------------
__global__ __launch_bounds__(256)
void ssd2_scan(float* __restrict__ states, const float* __restrict__ cdbuf) {
    int idx = blockIdx.x * 256 + threadIdx.x;          // h*8192 + p*128 + n
    if (idx >= NH * HD * SS) return;
    int h = idx >> 13;
    float carry = 0.f;
    for (int c = 0; c < NC; ++c) {
        size_t i = (size_t)c * (NH * HD * SS) + idx;
        float st = states[i];
        states[i] = carry;
        carry = carry * cdbuf[c * 128 + h] + st;
    }
}

// ---------------------------------------------------------------------------
// SSD stage 3: Yoff[l][p] = (C[l][n]*exp(cs[l])) @ prev[p][n]^T, accumulate.
// ---------------------------------------------------------------------------
__global__ __launch_bounds__(256)
void ssd3_kernel(const float* __restrict__ xBCs, const float* __restrict__ prev,
                 const float* __restrict__ csbuf, float* __restrict__ Ybuf) {
    __shared__ __bf16 sC[128 * 128];
    __shared__ __bf16 sP[64 * 128];
    __shared__ float s_ecs[128];
    const int c = blockIdx.x >> 7;
    const int h = blockIdx.x & 127;
    const int tid = threadIdx.x, lane = tid & 31, wave = tid >> 5;
    const int grp = h >> 4;

    if (tid < 128) s_ecs[tid] = __expf(csbuf[(size_t)(c * 128 + h) * 128 + tid]);
    __syncthreads();
    for (int i = tid; i < 128 * 128; i += 256) {
        int l = i >> 7, n = i & 127;
        sC[i] = (__bf16)(xBCs[(size_t)(c * 128 + l) * CONV_DIM + INTER + NG * SS
                              + grp * 128 + n] * s_ecs[l]);
    }
    for (int i = tid; i < 64 * 128; i += 256)
        sP[i] = (__bf16)prev[(size_t)(c * 128 + h) * (HD * SS) + i];
    __syncthreads();

    v8f acc[4];
#pragma unroll
    for (int j = 0; j < 4; ++j) acc[j] = (v8f){0.f,0.f,0.f,0.f,0.f,0.f,0.f,0.f};
    for (int k0 = 0; k0 < 128; k0 += 32) {
        v16bf a = ldsA(sC, 128, wave * 16, k0, lane);
#pragma unroll
        for (int j = 0; j < 4; ++j) {
            v16bf b = ldsB_rows(sP, 128, j * 16, k0, lane);  // B'[n][p] = prev[p][n]
            acc[j] = wmma_bf16(a, b, acc[j]);
        }
    }
#pragma unroll
    for (int j = 0; j < 4; ++j) {
        int p = j * 16 + (lane & 15);
#pragma unroll
        for (int r = 0; r < 8; ++r) {
            int l = wave * 16 + r + 8 * (lane >> 4);
            size_t gi = (size_t)(c * 128 + l) * INTER + h * 64 + p;
            Ybuf[gi] += acc[j][r];
        }
    }
}

// ---------------------------------------------------------------------------
// Gated group RMSNorm: out = (y*silu(gate)) * rstd(group) * norm_w
// ---------------------------------------------------------------------------
__global__ __launch_bounds__(256)
void gatenorm_kernel(const float* __restrict__ Ybuf, const float* __restrict__ proj,
                     const float* __restrict__ norm_w, float* __restrict__ normed) {
    __shared__ float red[8];
    const int t = blockIdx.x >> 3;
    const int g = blockIdx.x & 7;
    const int tid = threadIdx.x;
    size_t base = (size_t)t * INTER + g * 1024;
    size_t gbase = (size_t)t * PROJ + g * 1024;
    float vals[4];
    float ss = 0.f;
#pragma unroll
    for (int i = 0; i < 4; ++i) {
        int col = tid + i * 256;
        float gt = proj[gbase + col];
        float y = Ybuf[base + col] * (gt / (1.f + __expf(-gt)));
        vals[i] = y;
        ss += y * y;
    }
#pragma unroll
    for (int o = 16; o > 0; o >>= 1) ss += __shfl_down(ss, o, 32);
    if ((tid & 31) == 0) red[tid >> 5] = ss;
    __syncthreads();
    float tot = 0.f;
#pragma unroll
    for (int i = 0; i < 8; ++i) tot += red[i];
    float rstd = rsqrtf(tot * (1.f / 1024.f) + 1e-5f);
#pragma unroll
    for (int i = 0; i < 4; ++i) {
        int col = tid + i * 256;
        normed[base + col] = vals[i] * rstd * norm_w[g * 1024 + col];
    }
}

// ---------------------------------------------------------------------------
extern "C" void kernel_launch(void* const* d_in, const int* in_sizes, int n_in,
                              void* d_out, int out_size, void* d_ws, size_t ws_size,
                              hipStream_t stream) {
    const float* hs      = (const float*)d_in[0];
    const float* in_w    = (const float*)d_in[1];
    const float* conv_w  = (const float*)d_in[2];
    const float* conv_b  = (const float*)d_in[3];
    const float* dt_bias = (const float*)d_in[4];
    const float* Dp      = (const float*)d_in[5];
    const float* norm_w  = (const float*)d_in[6];
    const float* out_w   = (const float*)d_in[7];
    float* out = (float*)d_out;

    char* ws = (char*)d_ws;
    float* proj   = (float*)(ws);                    // 2048*18560*4 = 152,043,520
    float* xBCs   = (float*)(ws + 152043520);        // 2048*10240*4 =  83,886,080
    float* Ybuf   = (float*)(ws + 235929600);        // 2048*8192*4  =  67,108,864
    float* normed = (float*)(ws + 303038464);        // 2048*8192*4  =  67,108,864
    float* states = (float*)(ws + 370147328);        // 16*128*64*128*4 = 67,108,864
    float* csbuf  = (float*)(ws + 437256192);        // 16*128*128*4 =   1,048,576
    float* cdbuf  = (float*)(ws + 438304768);        // 16*128*4     =       8,192

    // 1. in_proj: proj = hs @ in_w^T  [2048,18560]; grid = (N/64)*4 stripes
    gemm_bf16w<<<(PROJ / 64) * 4, 256, 0, stream>>>(hs, in_w, proj, SEQ, PROJ, H_SIZE);
    // 2. causal depthwise conv + SiLU
    conv_silu_kernel<<<(SEQ * CONV_DIM) / 256, 256, 0, stream>>>(proj, conv_w, conv_b, xBCs);
    // 3. SSD intra-chunk (scores, Ydiag, chunk states)
    ssd1_kernel<<<NC * NH, 256, 0, stream>>>(proj, xBCs, dt_bias, Dp,
                                             Ybuf, states, csbuf, cdbuf);
    // 4. inter-chunk recurrence
    ssd2_scan<<<(NH * HD * SS) / 256, 256, 0, stream>>>(states, cdbuf);
    // 5. SSD inter-chunk contribution
    ssd3_kernel<<<NC * NH, 256, 0, stream>>>(xBCs, states, csbuf, Ybuf);
    // 6. gated group RMSNorm
    gatenorm_kernel<<<SEQ * NG, 256, 0, stream>>>(Ybuf, proj, norm_w, normed);
    // 7. out_proj: out = normed @ out_w^T  [2048,4096]
    gemm_bf16w<<<(H_SIZE / 64) * 4, 256, 0, stream>>>(normed, out_w, out, SEQ, H_SIZE, INTER);
}
